// Attention_11355893530823
// MI455X (gfx1250) — matrix operand order repaired and checked by reading, not compiled, not run on hardware
//
#include <hip/hip_runtime.h>

typedef __bf16 bf16_t;
typedef __attribute__((ext_vector_type(16))) __bf16 v16bf;
typedef __attribute__((ext_vector_type(8)))  float  v8f;
typedef __attribute__((vector_size(4 * sizeof(int)))) int v4i_vs;   // matches builtin param

#define HD 64
#define ED 32
#define WAVES_PER_BLOCK 4         // 128 threads; 48KB LDS/block (f32 tiles)
#define EDGES_PER_TILE 16

#define AS1 __attribute__((address_space(1)))
#define AS3 __attribute__((address_space(3)))

// ---------------------------------------------------------------------------
// CDNA5 async global->LDS gather (ASYNCcnt path), with sync fallback.
// ---------------------------------------------------------------------------
#if __has_builtin(__builtin_amdgcn_global_load_async_to_lds_b128)
#define HAVE_ASYNC_LDS 1
#else
#define HAVE_ASYNC_LDS 0
#endif

static __device__ __forceinline__ void wait_async0() {
#if __has_builtin(__builtin_amdgcn_s_wait_asynccnt)
    __builtin_amdgcn_s_wait_asynccnt(0);
#else
    asm volatile("s_wait_asynccnt 0x0" ::: "memory");
#endif
}

#if HAVE_ASYNC_LDS
static __device__ __forceinline__ void async_gather_b128(const float* g, float* lds) {
    // per-lane 64-bit global address, per-lane 32-bit LDS address
    __builtin_amdgcn_global_load_async_to_lds_b128(
        (AS1 v4i_vs*)(unsigned long long)(uintptr_t)g,
        (AS3 v4i_vs*)(unsigned)(uintptr_t)lds,
        0, 0);
}
#endif

// ---------------------------------------------------------------------------
// Kernel 0: zero the output accumulator and the per-node denominator.
// ---------------------------------------------------------------------------
__global__ __launch_bounds__(256) void zero_kernel(float* __restrict__ out, int n_out,
                                                   float* __restrict__ denom, int n_nodes) {
    int i = blockIdx.x * blockDim.x + threadIdx.x;
    if (i < n_out)   out[i]   = 0.0f;
    if (i < n_nodes) denom[i] = 0.0f;
}

// ---------------------------------------------------------------------------
// Kernel 1: per-edge attention logits via WMMA.
// Each wave owns 16 edges: gathers 16 q-rows / 16 k-rows (64-dim f32) and
// 16+16 eigs rows (32-dim f32) into LDS via async global->LDS b128 copies,
// converts to bf16 during fragment assembly, runs v_wmma_f32_16x16x32_bf16
// (2 K-chunks for q.k, 1 for eigs), takes the diagonal, and accumulates
// e = clip(exp(x/8 + exp(l0)*y), -5, 5) into denom[i0] with f32 atomics.
// ---------------------------------------------------------------------------
__global__ __launch_bounds__(128) void edge_logits_kernel(
    const float* __restrict__ q, const float* __restrict__ kmat,
    const float* __restrict__ eigs, const float* __restrict__ lambda0,
    const long long* __restrict__ idx,
    float* __restrict__ denom, float* __restrict__ e_scratch, int n_edges)
{
    __shared__ __align__(16) float fQ [WAVES_PER_BLOCK][EDGES_PER_TILE][HD];
    __shared__ __align__(16) float fK [WAVES_PER_BLOCK][EDGES_PER_TILE][HD];
    __shared__ __align__(16) float fE0[WAVES_PER_BLOCK][EDGES_PER_TILE][ED];
    __shared__ __align__(16) float fE1[WAVES_PER_BLOCK][EDGES_PER_TILE][ED];

    const int wave = threadIdx.x >> 5;
    const int lane = threadIdx.x & 31;
    const int tileBase = (blockIdx.x * WAVES_PER_BLOCK + wave) * EDGES_PER_TILE;

    // lanes 0..15 hold the edge indices for the 16 edges of this tile
    int i0 = 0, i1 = 0;
    if (lane < 16) {
        const int e = tileBase + lane;
        if (e < n_edges) {
            i0 = (int)idx[e];
            i1 = (int)idx[(size_t)n_edges + e];
        }
    }

    // ---- gather q/k rows: 16 rows x 64 f32 (b128 per lane per iter)
    #pragma unroll
    for (int it = 0; it < 8; ++it) {
        const int row = (lane >> 4) + it * 2;     // 0..15
        const int col = (lane & 15) * 4;          // 0..60, 16B aligned
        const int nq = __shfl(i0, row);
        const int nk = __shfl(i1, row);
        const float* gq = q    + (size_t)nq * HD + col;
        const float* gk = kmat + (size_t)nk * HD + col;
#if HAVE_ASYNC_LDS
        async_gather_b128(gq, &fQ[wave][row][col]);
        async_gather_b128(gk, &fK[wave][row][col]);
#else
        *(float4*)&fQ[wave][row][col] = *(const float4*)gq;
        *(float4*)&fK[wave][row][col] = *(const float4*)gk;
#endif
    }
    // ---- gather eigs rows: 16 rows x 32 f32, for both endpoints
    #pragma unroll
    for (int it = 0; it < 4; ++it) {
        const int row = (lane >> 3) + it * 4;     // 0..15
        const int col = (lane & 7) * 4;           // 0..28
        const int n0 = __shfl(i0, row);
        const int n1 = __shfl(i1, row);
        const float* g0 = eigs + (size_t)n0 * ED + col;
        const float* g1 = eigs + (size_t)n1 * ED + col;
#if HAVE_ASYNC_LDS
        async_gather_b128(g0, &fE0[wave][row][col]);
        async_gather_b128(g1, &fE1[wave][row][col]);
#else
        *(float4*)&fE0[wave][row][col] = *(const float4*)g0;
        *(float4*)&fE1[wave][row][col] = *(const float4*)g1;
#endif
    }
#if HAVE_ASYNC_LDS
    wait_async0();                // our wave's async LDS writes have landed
#endif
    __syncthreads();

    // ---- build WMMA fragments per ISA 16-bit layouts (convert f32 -> bf16).
    // A (16x32, M x K): lane<16 holds K 0..7 & 16..23; lane>=16 holds K 8..15 & 24..31.
    // B (32x16, K x N): same K order as A (lane<16: K 0..7 & 16..23; lane>=16: K 8..15 & 24..31); N = lane&15.
    const int m   = lane & 15;
    const int hiA = (lane >= 16) ? 8  : 0;
    const int hiB = (lane >= 16) ? 16 : 0;

    v8f acc  = {0.f, 0.f, 0.f, 0.f, 0.f, 0.f, 0.f, 0.f};
    #pragma unroll
    for (int c = 0; c < 2; ++c) {
        v16bf a, b;
        #pragma unroll
        for (int j = 0; j < 16; ++j) {
            const int ka = c * 32 + hiA + ((j < 8) ? j : (j + 8));
            const int kb = c * 32 + hiA + ((j < 8) ? j : (j + 8));
            a[j] = (bf16_t)fQ[wave][m][ka];
            b[j] = (bf16_t)fK[wave][m][kb];
        }
        acc = __builtin_amdgcn_wmma_f32_16x16x32_bf16(false, a, false, b,
                                                      (short)0, acc, false, false);
    }

    v8f accE = {0.f, 0.f, 0.f, 0.f, 0.f, 0.f, 0.f, 0.f};
    {
        v16bf a, b;
        #pragma unroll
        for (int j = 0; j < 16; ++j) {
            const int ka = hiA + ((j < 8) ? j : (j + 8));   // K = 0..31 (single chunk)
            const int kb = hiA + ((j < 8) ? j : (j + 8));
            a[j] = (bf16_t)fE0[wave][m][ka];
            b[j] = (bf16_t)fE1[wave][m][kb];
        }
        accE = __builtin_amdgcn_wmma_f32_16x16x32_bf16(false, a, false, b,
                                                       (short)0, accE, false, false);
    }

    // ---- extract the diagonal of the 16x16 accumulator.
    // C/D layout: lanes 0-15 -> N=lane, M=vgpr r; lanes 16-31 -> N=lane-16, M=8+r.
    // Diagonal lives in lanes 0-7 (elem=lane, edge=lane) and lanes 24-31
    // (elem=lane-24, edge=lane-16).
    const bool hasDiag = (lane < 8) || (lane >= 24);
    const int  elem = (lane < 8) ? lane : ((lane >= 24) ? (lane - 24) : 0);
    const int  slot = hasDiag ? ((lane < 8) ? lane : (lane - 16)) : 0;

    float x = 0.f, y = 0.f;
    #pragma unroll
    for (int r = 0; r < 8; ++r) {
        if (r == elem) { x = acc[r]; y = accE[r]; }
    }

    // broadcast i0 of this edge BEFORE divergence (bpermute reads need active src)
    const int d0 = __shfl(i0, slot);

    if (hasDiag) {
        const int e = tileBase + slot;
        if (e < n_edges) {
            const float expl = __expf(lambda0[0]);
            const float s  = x * 0.125f + expl * y;       // 1/sqrt(64) = 0.125
            float ev = __expf(s);
            ev = fminf(fmaxf(ev, -5.0f), 5.0f);           // jnp.clip semantics
            atomicAdd(denom + d0, ev);
            e_scratch[e] = ev;
        }
    }
}

// ---------------------------------------------------------------------------
// Kernel 2: SPMM scatter. 16 lanes per edge, each handles 4 of the 64 dims.
// Group leader loads the per-edge scalars once; broadcast within the group.
// out[i0] += (e / denom[i0]) * v[i1]
// ---------------------------------------------------------------------------
__global__ __launch_bounds__(256) void spmm_kernel(
    const float* __restrict__ v, const long long* __restrict__ idx,
    const float* __restrict__ denom, const float* __restrict__ e_scratch,
    float* __restrict__ out, int n_edges)
{
    const int t    = blockIdx.x * blockDim.x + threadIdx.x;
    const int edge = t >> 4;
    const int lane = threadIdx.x & 31;
    const int dim4 = (t & 15) * 4;
    if (edge >= n_edges) return;          // uniform within each 16-lane group

    int i0 = 0, i1 = 0;
    float w = 0.0f;
    if ((lane & 15) == 0) {               // one leader per 16-lane group
        i0 = (int)idx[edge];
        i1 = (int)idx[(size_t)n_edges + edge];
        float d = denom[i0];
        d = (d == 0.0f) ? 1.0f : d;
        w = e_scratch[edge] / d;
    }
    const int leader = lane & 16;         // 0 or 16
    i0 = __shfl(i0, leader);
    i1 = __shfl(i1, leader);
    w  = __shfl(w,  leader);

    const float4 vv = *(const float4*)(v + (size_t)i1 * HD + dim4);
    float* o = out + (size_t)i0 * HD + dim4;
    atomicAdd(o + 0, w * vv.x);
    atomicAdd(o + 1, w * vv.y);
    atomicAdd(o + 2, w * vv.z);
    atomicAdd(o + 3, w * vv.w);
}

// ---------------------------------------------------------------------------
extern "C" void kernel_launch(void* const* d_in, const int* in_sizes, int n_in,
                              void* d_out, int out_size, void* d_ws, size_t ws_size,
                              hipStream_t stream) {
    (void)n_in; (void)out_size; (void)ws_size;

    const float*     q       = (const float*)d_in[0];
    const float*     k       = (const float*)d_in[1];
    const float*     v       = (const float*)d_in[2];
    const float*     eigs    = (const float*)d_in[3];
    const float*     lambda0 = (const float*)d_in[4];
    const long long* idx     = (const long long*)d_in[5];   // int64 indices [2, E]

    const int n_nodes = in_sizes[0] / HD;
    const int n_edges = in_sizes[5] / 2;

    float* denom     = (float*)d_ws;            // [n_nodes]
    float* e_scratch = denom + n_nodes;         // [n_edges]
    float* out       = (float*)d_out;           // [n_nodes, 64]

    {   // zero out + denom
        const int n = n_nodes * HD;
        const int blocks = (n + 255) / 256;
        zero_kernel<<<blocks, 256, 0, stream>>>(out, n, denom, n_nodes);
    }
    {   // edge logits + denominator accumulation (WMMA path)
        const int tiles  = (n_edges + EDGES_PER_TILE - 1) / EDGES_PER_TILE;
        const int blocks = (tiles + WAVES_PER_BLOCK - 1) / WAVES_PER_BLOCK;
        edge_logits_kernel<<<blocks, 128, 0, stream>>>(q, k, eigs, lambda0, idx,
                                                       denom, e_scratch, n_edges);
    }
    {   // normalized scatter into the output
        const long long total  = (long long)n_edges * 16;
        const int       blocks = (int)((total + 255) / 256);
        spmm_kernel<<<blocks, 256, 0, stream>>>(v, idx, denom, e_scratch, out, n_edges);
    }
}
